// MaskedMultiHeadAttention_72954314490087
// MI455X (gfx1250) — matrix operand order repaired and checked by reading, not compiled, not run on hardware
//
#include <hip/hip_runtime.h>

typedef __attribute__((ext_vector_type(16))) _Float16 v16h;
typedef __attribute__((ext_vector_type(8)))  _Float16 v8h;
typedef __attribute__((ext_vector_type(8)))  float    v8f;

#define D_MODEL 1024
#define S_LEN   2048
#define NH      16
#define HD      64
#define N_TOTAL (3 * D_MODEL)   // 3072
#define M_TOTAL (2 * S_LEN)     // 4096
#define SCALE   0.03125f        // 1/sqrt(1024)

// ---- fragment helpers -------------------------------------------------------
// 16-bit A-matrix 16x32 layout (contiguous, matches B): lane m (low half)  holds K=0..15
//                                                       lane m (high half) holds K=16..31
__device__ __forceinline__ v16h a_frag_from_f32(const float* __restrict__ row, int hi) {
    const float* p0 = row + hi * 16;
    const float* p1 = row + hi * 16 + 8;
    v16h r;
#pragma unroll
    for (int i = 0; i < 8; ++i) r[i]     = (_Float16)p0[i];
#pragma unroll
    for (int i = 0; i < 8; ++i) r[i + 8] = (_Float16)p1[i];
    return r;
}

// 16-bit B-matrix 32x16: lane n (low half) holds K=0..15, high half K=16..31 (contiguous)
__device__ __forceinline__ v16h b_frag_from_f32(const float* __restrict__ p) {
    v16h r;
#pragma unroll
    for (int i = 0; i < 16; ++i) r[i] = (_Float16)p[i];
    return r;
}

// f16 A-fragment from two 16-byte-aligned chunks
__device__ __forceinline__ v16h a_frag_h(const _Float16* p0, const _Float16* p1) {
    const v8h a = *(const v8h*)p0;
    const v8h b = *(const v8h*)p1;
    v16h r;
#pragma unroll
    for (int i = 0; i < 8; ++i) { r[i] = a[i]; r[i + 8] = b[i]; }
    return r;
}

__device__ __forceinline__ v8f wmma_f16(v16h a, v16h b, v8f c) {
    return __builtin_amdgcn_wmma_f32_16x16x32_f16(false, a, false, b, (short)0, c,
                                                  false, false);
}

// ---- Kernel 1: QKV projection (z = x*W^T + b), staged to f16 Q/K/Vt ---------
// 32x64 output tile per wave: 2 A-frags x 4 B-frags -> 8 back-to-back WMMAs/K-step.
__global__ __launch_bounds__(32)
void qkv_gemm_kernel(const float* __restrict__ x, const float* __restrict__ W,
                     const float* __restrict__ bias,
                     _Float16* __restrict__ Qh, _Float16* __restrict__ Kh,
                     _Float16* __restrict__ Vt) {
    const int lane = threadIdx.x;
    const int m  = lane & 15;
    const int hi = lane >> 4;
    const int nbase = blockIdx.x * 64;   // 64 output cols (4 N-subtiles)
    const int mbase = blockIdx.y * 32;   // 32 output rows (2 M-subtiles)

    const float* arow0 = x + (size_t)(mbase + m) * D_MODEL;
    const float* arow1 = x + (size_t)(mbase + 16 + m) * D_MODEL;
    const float* brow0 = W + (size_t)(nbase +  0 + m) * D_MODEL + hi * 16;
    const float* brow1 = W + (size_t)(nbase + 16 + m) * D_MODEL + hi * 16;
    const float* brow2 = W + (size_t)(nbase + 32 + m) * D_MODEL + hi * 16;
    const float* brow3 = W + (size_t)(nbase + 48 + m) * D_MODEL + hi * 16;

    v8f c[2][4];
#pragma unroll
    for (int u = 0; u < 2; ++u)
#pragma unroll
        for (int j = 0; j < 4; ++j)
#pragma unroll
            for (int i = 0; i < 8; ++i) c[u][j][i] = 0.0f;

    for (int kk = 0; kk < D_MODEL; kk += 32) {
        // Load ALL fragments into distinct registers, then burst 8 WMMAs
        // (independent back-to-back WMMAs need no hazard nops).
        const v16h a0 = a_frag_from_f32(arow0 + kk, hi);
        const v16h a1 = a_frag_from_f32(arow1 + kk, hi);
        const v16h b0 = b_frag_from_f32(brow0 + kk);
        const v16h b1 = b_frag_from_f32(brow1 + kk);
        const v16h b2 = b_frag_from_f32(brow2 + kk);
        const v16h b3 = b_frag_from_f32(brow3 + kk);
        if (kk + 32 < D_MODEL) {
            __builtin_prefetch(arow0 + kk + 32, 0, 0);   // global_prefetch_b8
            __builtin_prefetch(arow1 + kk + 32, 0, 0);
        }
        c[0][0] = wmma_f16(a0, b0, c[0][0]);
        c[1][0] = wmma_f16(a1, b0, c[1][0]);
        c[0][1] = wmma_f16(a0, b1, c[0][1]);
        c[1][1] = wmma_f16(a1, b1, c[1][1]);
        c[0][2] = wmma_f16(a0, b2, c[0][2]);
        c[1][2] = wmma_f16(a1, b2, c[1][2]);
        c[0][3] = wmma_f16(a0, b3, c[0][3]);
        c[1][3] = wmma_f16(a1, b3, c[1][3]);
    }

    // Epilogue: C layout = VGPR r <-> row r+8*hi, lane%16 <-> col
#pragma unroll
    for (int j = 0; j < 4; ++j) {
        const int col    = nbase + j * 16 + m;
        const int region = col >> 10;       // 0:k 1:q 2:v  (uniform per 16-col tile)
        const int d      = col & 1023;
        const int h      = d >> 6;
        const int hd     = d & 63;
        const float bv   = bias[col];
#pragma unroll
        for (int u = 0; u < 2; ++u)
#pragma unroll
            for (int r = 0; r < 8; ++r) {
                const int row = mbase + u * 16 + r + 8 * hi;
                const int n   = row >> 11;
                const int s   = row & 2047;
                const _Float16 hv = (_Float16)(c[u][j][r] + bv);
                const size_t qk = (((size_t)n * NH + h) * S_LEN + s) * HD + hd;
                if (region == 0)      Kh[qk] = hv;
                else if (region == 1) Qh[qk] = hv;
                else                  Vt[(((size_t)n * NH + h) * HD + hd) * S_LEN + s] = hv;
            }
    }
}

// ---- Kernel 2: causal flash attention, one wave per (n,h,32-query tile) ----
// 32-query tile reuses each K/V fragment across two 16-query sub-tiles.
__global__ __launch_bounds__(32)
void attn_kernel(const _Float16* __restrict__ Qh, const _Float16* __restrict__ Kh,
                 const _Float16* __restrict__ Vt, float* __restrict__ out) {
    __shared__ alignas(32) _Float16 Pb[32 * 32];

    const int lane = threadIdx.x;
    const int m  = lane & 15;
    const int hi = lane >> 4;
    const int q0 = blockIdx.x * 32;
    const int h  = blockIdx.y;
    const int n  = blockIdx.z;

    const _Float16* Qb = Qh + ((size_t)(n * NH + h)) * S_LEN * HD;
    const _Float16* Kb = Kh + ((size_t)(n * NH + h)) * S_LEN * HD;
    const _Float16* Vb = Vt + ((size_t)(n * NH + h)) * HD * S_LEN;

    // Q A-fragments for both 16-row sub-tiles, resident for the whole loop.
    v16h qa[2][2];
#pragma unroll
    for (int u = 0; u < 2; ++u) {
        const _Float16* qrow = Qb + (size_t)(q0 + u * 16 + m) * HD;
        qa[u][0] = a_frag_h(qrow + hi * 16,      qrow + hi * 16 + 8);
        qa[u][1] = a_frag_h(qrow + 32 + hi * 16, qrow + 32 + hi * 16 + 8);
    }

    float mrow[2][8], lrow[2][8];
    v8f o[2][4];
#pragma unroll
    for (int u = 0; u < 2; ++u) {
#pragma unroll
        for (int r = 0; r < 8; ++r) { mrow[u][r] = -3.0e38f; lrow[u][r] = 0.0f; }
#pragma unroll
        for (int j = 0; j < 4; ++j)
#pragma unroll
            for (int i = 0; i < 8; ++i) o[u][j][i] = 0.0f;
    }

    for (int kb = 0; kb < q0 + 32; kb += 32) {
        // ---- scores: K B-fragments loaded once, used by both query sub-tiles ----
        v8f sc[2][2];   // [u][t]
#pragma unroll
        for (int t = 0; t < 2; ++t) {
            const _Float16* krow = Kb + (size_t)(kb + t * 16 + m) * HD;
            const v16h b0 = *(const v16h*)(krow + hi * 16);        // hd 0..31
            const v16h b1 = *(const v16h*)(krow + 32 + hi * 16);   // hd 32..63
#pragma unroll
            for (int u = 0; u < 2; ++u) {
                v8f z;
#pragma unroll
                for (int i = 0; i < 8; ++i) z[i] = 0.0f;
                z = wmma_f16(qa[u][0], b0, z);
                z = wmma_f16(qa[u][1], b1, z);
                sc[u][t] = z;
            }
        }

        // ---- causal mask + online softmax (row u*16+r+8*hi, col m) ----
        float p[2][2][8];
#pragma unroll
        for (int u = 0; u < 2; ++u) {
            float bm[8];
#pragma unroll
            for (int r = 0; r < 8; ++r) {
                const int q = q0 + u * 16 + r + 8 * hi;
                float s0 = sc[u][0][r] * SCALE;
                float s1 = sc[u][1][r] * SCALE;
                if (kb + m > q)      s0 = -1.0e30f;
                if (kb + 16 + m > q) s1 = -1.0e30f;
                sc[u][0][r] = s0; sc[u][1][r] = s1;
                bm[r] = fmaxf(s0, s1);
            }
#pragma unroll
            for (int msk = 1; msk < 16; msk <<= 1)
#pragma unroll
                for (int r = 0; r < 8; ++r)
                    bm[r] = fmaxf(bm[r], __shfl_xor(bm[r], msk, 32));

            float rs[8];
#pragma unroll
            for (int r = 0; r < 8; ++r) {
                const float mn = fmaxf(mrow[u][r], bm[r]);
                const float alpha = __expf(mrow[u][r] - mn);
                const float p0 = __expf(sc[u][0][r] - mn);
                const float p1 = __expf(sc[u][1][r] - mn);
                p[u][0][r] = p0; p[u][1][r] = p1;
                rs[r] = p0 + p1;
                mrow[u][r] = mn;
                lrow[u][r] *= alpha;
#pragma unroll
                for (int j = 0; j < 4; ++j) o[u][j][r] *= alpha;
            }
#pragma unroll
            for (int msk = 1; msk < 16; msk <<= 1)
#pragma unroll
                for (int r = 0; r < 8; ++r)
                    rs[r] += __shfl_xor(rs[r], msk, 32);
#pragma unroll
            for (int r = 0; r < 8; ++r) lrow[u][r] += rs[r];
        }

        // ---- P: C layout -> A layout via LDS ----
#pragma unroll
        for (int u = 0; u < 2; ++u)
#pragma unroll
            for (int t = 0; t < 2; ++t)
#pragma unroll
                for (int r = 0; r < 8; ++r)
                    Pb[(u * 16 + r + 8 * hi) * 32 + t * 16 + m] = (_Float16)p[u][t][r];
        __syncthreads();
        v16h pa[2];
#pragma unroll
        for (int u = 0; u < 2; ++u) {
            const _Float16* prow = &Pb[(u * 16 + m) * 32];
            pa[u] = a_frag_h(prow + hi * 16, prow + hi * 16 + 8);
        }

        // ---- O += P * V: V B-fragments loaded once, used by both sub-tiles ----
#pragma unroll
        for (int j = 0; j < 4; ++j) {
            const v16h vb =
                *(const v16h*)(Vb + (size_t)(j * 16 + m) * S_LEN + kb + hi * 16);
            o[0][j] = wmma_f16(pa[0], vb, o[0][j]);
            o[1][j] = wmma_f16(pa[1], vb, o[1][j]);
        }
        __syncthreads();
    }

    // ---- normalize + store fp32 output [n, q, h*64 + hd] ----
#pragma unroll
    for (int u = 0; u < 2; ++u)
#pragma unroll
        for (int j = 0; j < 4; ++j)
#pragma unroll
            for (int r = 0; r < 8; ++r) {
                const int q = q0 + u * 16 + r + 8 * hi;
                out[((size_t)n * S_LEN + q) * D_MODEL + h * HD + j * 16 + m] =
                    o[u][j][r] / lrow[u][r];
            }
}

// ---- launcher ---------------------------------------------------------------
extern "C" void kernel_launch(void* const* d_in, const int* in_sizes, int n_in,
                              void* d_out, int out_size, void* d_ws, size_t ws_size,
                              hipStream_t stream) {
    (void)in_sizes; (void)n_in; (void)out_size; (void)ws_size;
    const float* x    = (const float*)d_in[0];
    const float* W    = (const float*)d_in[1];
    const float* bias = (const float*)d_in[2];
    float* out        = (float*)d_out;

    _Float16* ws = (_Float16*)d_ws;
    const size_t per = (size_t)2 * NH * S_LEN * HD;  // 4M halves = 8 MB each
    _Float16* Qh = ws;
    _Float16* Kh = ws + per;
    _Float16* Vt = ws + 2 * per;

    dim3 g1(N_TOTAL / 64, M_TOTAL / 32, 1);          // 48 x 128 waves
    qkv_gemm_kernel<<<g1, 32, 0, stream>>>(x, W, bias, Qh, Kh, Vt);

    dim3 g2(S_LEN / 32, NH, 2);                      // 64 x 16 x 2 waves
    attn_kernel<<<g2, 32, 0, stream>>>(Qh, Kh, Vt, out);
}
